// PCDAlignment_8770323219142
// MI455X (gfx1250) — matrix operand order repaired and checked by reading, not compiled, not run on hardware
//
#include <hip/hip_runtime.h>
#include <math.h>

// ---------------------------------------------------------------------------
// PCD alignment (EDVR) for MI455X / gfx1250.
// All convs + the deformable-conv einsum are lowered to bf16 WMMA GEMMs
// (V_WMMA_F32_16X16X32_BF16, f32 accumulate), register-blocked 2Mx4N per
// wave with the weight panel staged in LDS (bank-conflict-free, padded
// stride). Intermediates are bf16 channels-last; working set ~163MB targets
// the 192MB L2.
// ---------------------------------------------------------------------------

typedef unsigned short u16;
typedef unsigned int   u32;
typedef __bf16 v16bf __attribute__((ext_vector_type(16)));
typedef float  v8f   __attribute__((ext_vector_type(8)));

#define MC   64
#define NG   8
#define KTAP 9
#define NB   2

__device__ __forceinline__ float bf2f(u16 h) {
    return __uint_as_float(((u32)h) << 16);
}
__device__ __forceinline__ u16 f2bf(float f) {
    u32 u = __float_as_uint(f);
    u32 r = u + 0x7FFFu + ((u >> 16) & 1u);   // round-to-nearest-even
    return (u16)(r >> 16);
}

// ---------------------------------------------------------------------------
// f32 NCHW -> bf16 channels-last [N*H*W, C]
// ---------------------------------------------------------------------------
__global__ __launch_bounds__(256) void k_nchw2cl(const float* __restrict__ src,
                                                 u16* __restrict__ dst,
                                                 int n_total, int C, int HW) {
    int idx = blockIdx.x * blockDim.x + threadIdx.x;
    if (idx >= n_total) return;                       // n_total = N*C*HW
    int hw = idx % HW;
    int c  = (idx / HW) % C;
    int n  = idx / (HW * C);
    dst[((size_t)(n * HW + hw)) * C + c] = f2bf(src[idx]);
}

// ---------------------------------------------------------------------------
// OIHW f32 weights -> bf16 B matrix [CoutPad][Cin*9]
//   tap_major=1: k = tap*Cin + ci   (conv im2col order)
//   tap_major=0: k = ci*9 + tap     (deformable einsum order)
// ---------------------------------------------------------------------------
__global__ __launch_bounds__(256) void k_wxform(const float* __restrict__ w,
                                                int Cout, int Cin, int CoutPad,
                                                int tap_major,
                                                u16* __restrict__ B) {
    int idx = blockIdx.x * blockDim.x + threadIdx.x;
    int Kt = Cin * 9;
    if (idx >= CoutPad * Kt) return;
    int k  = idx % Kt;
    int co = idx / Kt;
    u16 v = 0;
    if (co < Cout) {
        int ci, tap;
        if (tap_major) { tap = k / Cin; ci = k % Cin; }
        else           { ci  = k / 9;   tap = k % 9;  }
        v = f2bf(w[((size_t)co * Cin + ci) * 9 + tap]);
    }
    B[(size_t)co * Kt + k] = v;
}

// ---------------------------------------------------------------------------
// im2col (zero-pad SAME) with optional channel-concat of two tensors.
// A[p][tap*C + ci], C = C1+C2, bf16. 16B-vector copies (C1,C2 multiples of 8).
// ---------------------------------------------------------------------------
__global__ __launch_bounds__(256) void k_im2col(const u16* __restrict__ in1,
                                                const u16* __restrict__ in2,
                                                int C1, int C2,
                                                int Nn, int H, int W,
                                                u16* __restrict__ A) {
    int idx = blockIdx.x * blockDim.x + threadIdx.x;   // (p, tap)
    int total = Nn * H * W * 9;
    if (idx >= total) return;
    int tap = idx % 9;
    int p   = idx / 9;
    int x = p % W;
    int y = (p / W) % H;
    int n = p / (W * H);
    int yy = y + tap / 3 - 1;
    int xx = x + tap % 3 - 1;
    int C = C1 + C2;
    u16* dst = A + (size_t)p * 9 * C + (size_t)tap * C;
    if (yy < 0 || yy >= H || xx < 0 || xx >= W) {
        uint4 z = make_uint4(0u, 0u, 0u, 0u);
        for (int c = 0; c < C; c += 8) *(uint4*)(dst + c) = z;
    } else {
        size_t sp = ((size_t)n * H + yy) * W + xx;
        const u16* s1 = in1 + sp * C1;
        for (int c = 0; c < C1; c += 8) *(uint4*)(dst + c) = *(const uint4*)(s1 + c);
        if (C2) {
            const u16* s2 = in2 + sp * C2;
            for (int c = 0; c < C2; c += 8) *(uint4*)(dst + C1 + c) = *(const uint4*)(s2 + c);
        }
    }
}

// ---------------------------------------------------------------------------
// WMMA GEMM: D[p,co] = act( sum_k A[p,k]*B[co,k] + bias[co] )
//
// Block = 128 threads (4 waves). Each block owns a 64-wide Cout group whose
// bf16 weight panel [64 x Ktot] is cooperatively staged into LDS with a
// +8-half padded row stride (row stride mod 64 dwords = 4 -> b128 reads are
// bank-conflict free). Each wave computes 32 pixels x 64 Cout (2Mx4N tiles):
// per k-step, 2 A-fragments (global b128) + 4 B-fragments (ds_load_b128,
// preloaded into distinct registers so the WMMA burst isn't serialized by
// WAR waits) feed 8 WMMAs -> 24 fragment-bytes per lane per WMMA.
// outmode: 0 = bf16 channels-last, 1 = f32 channels-last, 2 = f32 NCHW.
// ---------------------------------------------------------------------------
__global__ __launch_bounds__(128) void k_gemm(const u16* __restrict__ A,
                                              const u16* __restrict__ B,
                                              const float* __restrict__ bias,
                                              void* __restrict__ out,
                                              int P, int Ktot, int Cout, int CoutPad,
                                              int HW, int outmode, int act) {
    extern __shared__ u16 sB[];
    const int KtotP = Ktot + 8;                 // padded LDS row stride (halves)
    const int n_groups = CoutPad >> 6;
    const int ngi = blockIdx.x % n_groups;
    const int bm  = blockIdx.x / n_groups;
    const int coBase = ngi << 6;

    // ---- cooperative stage of the 64-row B panel into LDS ----
    {
        const int rowVecs = Ktot >> 3;          // uint4 per row
        const int totalVecs = rowVecs << 6;     // 64 rows
        const uint4* srcB = (const uint4*)(B + (size_t)coBase * Ktot);
        for (int i = threadIdx.x; i < totalVecs; i += 128) {
            int r = i / rowVecs;
            int c = i - r * rowVecs;
            *(uint4*)(sB + (size_t)r * KtotP + (c << 3)) = srcB[i];
        }
    }
    __syncthreads();

    const int wv   = threadIdx.x >> 5;
    const int lane = threadIdx.x & 31;
    const int m16  = lane & 15;
    const int hi   = lane >> 4;                 // 0 or 1
    const int p0   = (bm * 4 + wv) * 32;        // 32-pixel strip for this wave

    const u16* pa0 = A + (size_t)(p0 + m16) * Ktot + hi * 8;        // A: K runs of 8
    const u16* pa1 = A + (size_t)(p0 + 16 + m16) * Ktot + hi * 8;
    const u16* pb  = sB + (size_t)m16 * KtotP + hi * 8;             // B: K runs of 8

    v8f acc[2][4];
#pragma unroll
    for (int mi = 0; mi < 2; ++mi)
#pragma unroll
        for (int j = 0; j < 4; ++j) {
            v8f z = {0.f, 0.f, 0.f, 0.f, 0.f, 0.f, 0.f, 0.f};
            acc[mi][j] = z;
        }

    for (int kk = 0; kk < Ktot; kk += 32) {
        union F16 { uint4 q[2]; v16bf v; };
        F16 a0, a1, b[4];
        a0.q[0] = *(const uint4*)(pa0 + kk);        // K = base+0..7
        a0.q[1] = *(const uint4*)(pa0 + kk + 16);   // K = base+16..23
        a1.q[0] = *(const uint4*)(pa1 + kk);
        a1.q[1] = *(const uint4*)(pa1 + kk + 16);
#pragma unroll
        for (int j = 0; j < 4; ++j) {               // preload all B frags (one ds clause)
            const u16* bj = pb + (size_t)(j << 4) * KtotP;
            b[j].q[0] = *(const uint4*)(bj + kk);   // ds_load_b128
            b[j].q[1] = *(const uint4*)(bj + kk + 16);
        }
        __builtin_prefetch(pa0 + kk + 512, 0, 3);   // near-scope A-stream prefetch
        __builtin_prefetch(pa1 + kk + 512, 0, 3);
#pragma unroll
        for (int j = 0; j < 4; ++j) {
            acc[0][j] = __builtin_amdgcn_wmma_f32_16x16x32_bf16(
                false, a0.v, false, b[j].v, (short)0, acc[0][j], false, false);
            acc[1][j] = __builtin_amdgcn_wmma_f32_16x16x32_bf16(
                false, a1.v, false, b[j].v, (short)0, acc[1][j], false, false);
        }
    }

    // ---- epilogue: bias + leaky-ReLU + store ----
#pragma unroll
    for (int j = 0; j < 4; ++j) {
        int co = coBase + (j << 4) + m16;
        bool cok = co < Cout;
        float bv = cok ? bias[co] : 0.f;
#pragma unroll
        for (int mi = 0; mi < 2; ++mi) {
#pragma unroll
            for (int r = 0; r < 8; ++r) {
                int p = p0 + (mi << 4) + r + hi * 8;   // C/D layout: VGPR r -> M=r (+8 hi)
                float v = acc[mi][j][r] + bv;
                if (act) v = (v >= 0.f) ? v : 0.1f * v;
                if (cok) {
                    if (outmode == 0) {
                        ((u16*)out)[(size_t)p * Cout + co] = f2bf(v);
                    } else if (outmode == 1) {
                        ((float*)out)[(size_t)p * Cout + co] = v;
                    } else {
                        int n = p / HW, hw = p % HW;
                        ((float*)out)[((size_t)n * Cout + co) * HW + hw] = v;
                    }
                }
            }
        }
    }
}

// ---------------------------------------------------------------------------
// Deformable sampling: builds cols-A [p][(g*8+c)*9 + tap] (bf16, mask folded)
// from bf16-CL features and the 216-ch f32-CL offset/mask conv output.
// ---------------------------------------------------------------------------
__global__ __launch_bounds__(256) void k_cols(const u16* __restrict__ xcl,
                                              const float* __restrict__ dout,
                                              int Nn, int H, int W,
                                              u16* __restrict__ A) {
    int idx = blockIdx.x * blockDim.x + threadIdx.x;   // (p, g, tap)
    int total = Nn * H * W * NG * KTAP;
    if (idx >= total) return;
    int tap = idx % KTAP;
    int g   = (idx / KTAP) % NG;
    int p   = idx / (KTAP * NG);
    int x = p % W;
    int y = (p / W) % H;
    int n = p / (W * H);

    const float* o = dout + (size_t)p * 216;
    float offy = o[(g * KTAP + tap) * 2 + 0];
    float offx = o[(g * KTAP + tap) * 2 + 1];
    float mask = 1.f / (1.f + expf(-o[144 + g * KTAP + tap]));

    float py = offy + (float)y + (float)(tap / 3 - 1);
    float px = offx + (float)x + (float)(tap % 3 - 1);
    float y0f = floorf(py), x0f = floorf(px);
    int   y0  = (int)y0f,   x0  = (int)x0f;
    float ly = py - y0f, lx = px - x0f;

    float a[8] = {0.f, 0.f, 0.f, 0.f, 0.f, 0.f, 0.f, 0.f};
    int cbase = g * 8;
    auto gather = [&](int yy, int xx, float wt) {
        if (yy < 0 || yy >= H || xx < 0 || xx >= W) return;
        const u16* s = xcl + (((size_t)n * H + yy) * W + xx) * MC + cbase;
        uint4 q = *(const uint4*)s;
        const u16* us = (const u16*)&q;
#pragma unroll
        for (int c = 0; c < 8; ++c) a[c] += bf2f(us[c]) * wt;
    };
    gather(y0,     x0,     (1.f - ly) * (1.f - lx));
    gather(y0,     x0 + 1, (1.f - ly) * lx);
    gather(y0 + 1, x0,     ly * (1.f - lx));
    gather(y0 + 1, x0 + 1, ly * lx);

    u16* dst = A + (size_t)p * 576 + (size_t)cbase * 9 + tap;  // k=(g*8+c)*9+tap
#pragma unroll
    for (int c = 0; c < 8; ++c) dst[c * 9] = f2bf(a[c] * mask);
}

// ---------------------------------------------------------------------------
// Bilinear x2 upsample (half-pixel centers, edge clamp), optional scale.
// bf16-CL src [N,h,w,C] -> dst [N,2h,2w,C].
// ---------------------------------------------------------------------------
__global__ __launch_bounds__(256) void k_up2(const u16* __restrict__ src,
                                             u16* __restrict__ dst,
                                             int Nn, int h, int w, int C,
                                             float scale) {
    int idx = blockIdx.x * blockDim.x + threadIdx.x;
    int total = Nn * 4 * h * w * C;
    if (idx >= total) return;
    int c = idx % C;
    int t = idx / C;
    int x = t % (2 * w);
    int y = (t / (2 * w)) % (2 * h);
    int n = t / (4 * h * w);
    float sy = (y + 0.5f) * 0.5f - 0.5f;
    float sx = (x + 0.5f) * 0.5f - 0.5f;
    float y0f = floorf(sy), x0f = floorf(sx);
    float ly = sy - y0f, lx = sx - x0f;
    int y0 = (int)y0f, x0 = (int)x0f;
    int y0c = min(max(y0, 0), h - 1),     y1c = min(max(y0 + 1, 0), h - 1);
    int x0c = min(max(x0, 0), w - 1),     x1c = min(max(x0 + 1, 0), w - 1);
    auto S = [&](int yy, int xx) {
        return bf2f(src[(((size_t)n * h + yy) * w + xx) * C + c]);
    };
    float v = (1.f - ly) * ((1.f - lx) * S(y0c, x0c) + lx * S(y0c, x1c)) +
              ly         * ((1.f - lx) * S(y1c, x0c) + lx * S(y1c, x1c));
    dst[idx] = f2bf(v * scale);
}

// ---------------------------------------------------------------------------
// Host orchestration
// ---------------------------------------------------------------------------
namespace {
struct Arena {
    char*  base;
    size_t off;
    void* alloc(size_t bytes) {
        size_t a = (off + 255) & ~(size_t)255;
        off = a + bytes;
        return base + a;
    }
};
}  // namespace

extern "C" void kernel_launch(void* const* d_in, const int* in_sizes, int n_in,
                              void* d_out, int out_size, void* d_ws, size_t ws_size,
                              hipStream_t stream) {
    (void)in_sizes; (void)n_in; (void)out_size; (void)ws_size;

    // ---- inputs (setup_inputs order) ----
    const float* nf1 = (const float*)d_in[0];
    const float* nf2 = (const float*)d_in[1];
    const float* nf3 = (const float*)d_in[2];
    const float* rf1 = (const float*)d_in[3];
    const float* rf2 = (const float*)d_in[4];
    const float* rf3 = (const float*)d_in[5];

    // ---- params: JAX pytree flattening = sorted dict keys ----
    const float* cas1_b = (const float*)d_in[6];
    const float* cas1_w = (const float*)d_in[7];
    const float* cas2_b = (const float*)d_in[8];
    const float* cas2_w = (const float*)d_in[9];
    const float* cas_b  = (const float*)d_in[10];
    const float* cas_ob = (const float*)d_in[11];
    const float* cas_ow = (const float*)d_in[12];
    const float* cas_w  = (const float*)d_in[13];
    const float* dcn_b[3]  = {(const float*)d_in[14], (const float*)d_in[15], (const float*)d_in[16]};
    const float* dcn_ob[3] = {(const float*)d_in[17], (const float*)d_in[18], (const float*)d_in[19]};
    const float* dcn_ow[3] = {(const float*)d_in[20], (const float*)d_in[21], (const float*)d_in[22]};
    const float* dcn_w[3]  = {(const float*)d_in[23], (const float*)d_in[24], (const float*)d_in[25]};
    const float* feat_b[2] = {(const float*)d_in[26], (const float*)d_in[27]};
    const float* feat_w[2] = {(const float*)d_in[28], (const float*)d_in[29]};
    const float* oc1_b[3]  = {(const float*)d_in[30], (const float*)d_in[31], (const float*)d_in[32]};
    const float* oc1_w[3]  = {(const float*)d_in[33], (const float*)d_in[34], (const float*)d_in[35]};
    const float* oc2_b[3]  = {(const float*)d_in[36], (const float*)d_in[37], (const float*)d_in[38]};
    const float* oc2_w[3]  = {(const float*)d_in[39], (const float*)d_in[40], (const float*)d_in[41]};
    const float* oc3_b[2]  = {(const float*)d_in[42], (const float*)d_in[43]};
    const float* oc3_w[2]  = {(const float*)d_in[44], (const float*)d_in[45]};

    const int h1 = 128, w1 = 128, h2 = 64, w2 = 64, h3 = 32, w3 = 32;
    const int P1 = NB * h1 * w1, P2 = NB * h2 * w2, P3 = NB * h3 * w3;

    Arena ar{(char*)d_ws, 0};
    auto abuf = [&](int P) { return (u16*)ar.alloc((size_t)P * MC * sizeof(u16)); };

    // scratch buffers
    u16* Aim      = (u16*)ar.alloc((size_t)P1 * 9 * 2 * MC * sizeof(u16)); // max im2col (K=1152)
    float* dcnbuf = (float*)ar.alloc((size_t)P1 * 216 * sizeof(float));    // offset/mask conv out

    u16 *cnf1 = abuf(P1), *crf1 = abuf(P1);
    u16 *cnf2 = abuf(P2), *crf2 = abuf(P2);
    u16 *cnf3 = abuf(P3), *crf3 = abuf(P3);
    u16 *tmp3 = abuf(P3), *off3 = abuf(P3), *feat3 = abuf(P3);
    u16 *upoff3 = abuf(P2), *upfeat3 = abuf(P2);
    u16 *t2a = abuf(P2), *t2b = abuf(P2), *off2 = abuf(P2), *featd2 = abuf(P2), *feat2 = abuf(P2);
    u16 *upoff2 = abuf(P1), *upfeat2 = abuf(P1);
    u16 *t1a = abuf(P1), *t1b = abuf(P1), *off1 = abuf(P1), *featd1 = abuf(P1), *feat1 = abuf(P1);
    u16 *tca = abuf(P1), *tcb = abuf(P1);

    // ---- weight transforms ----
    auto xform = [&](const float* w, int Cout, int Cin, int CoutPad, int tap_major) {
        u16* B = (u16*)ar.alloc((size_t)CoutPad * Cin * 9 * sizeof(u16));
        int total = CoutPad * Cin * 9;
        k_wxform<<<(total + 255) / 256, 256, 0, stream>>>(w, Cout, Cin, CoutPad, tap_major, B);
        return B;
    };
    u16* B_oc1[3]; u16* B_oc2[3]; u16* B_oc3[2]; u16* B_feat[2];
    u16* B_dow[3]; u16* B_dw[3];
    for (int i = 0; i < 3; ++i) {
        B_oc1[i] = xform(oc1_w[i], MC, 2 * MC, MC, 1);
        B_oc2[i] = xform(oc2_w[i], MC, (i == 2) ? MC : 2 * MC, MC, 1);  // l3 has Cin=MC
        B_dow[i] = xform(dcn_ow[i], 216, MC, 256, 1);                   // pad 216 -> 256
        B_dw[i]  = xform(dcn_w[i],  MC,  MC, MC, 0);                    // einsum order
    }
    for (int i = 0; i < 2; ++i) {
        B_oc3[i]  = xform(oc3_w[i],  MC, MC,     MC, 1);
        B_feat[i] = xform(feat_w[i], MC, 2 * MC, MC, 1);
    }
    u16* B_cas1 = xform(cas1_w, MC, 2 * MC, MC, 1);
    u16* B_cas2 = xform(cas2_w, MC, MC,     MC, 1);
    u16* B_cow  = xform(cas_ow, 216, MC,    256, 1);
    u16* B_cw   = xform(cas_w,  MC, MC,     MC, 0);

    // ---- input conversion ----
    auto cvt = [&](const float* s, u16* d, int P, int HW) {
        int total = P * MC;
        k_nchw2cl<<<(total + 255) / 256, 256, 0, stream>>>(s, d, total, MC, HW);
    };
    cvt(nf1, cnf1, P1, h1 * w1); cvt(rf1, crf1, P1, h1 * w1);
    cvt(nf2, cnf2, P2, h2 * w2); cvt(rf2, crf2, P2, h2 * w2);
    cvt(nf3, cnf3, P3, h3 * w3); cvt(rf3, crf3, P3, h3 * w3);

    // ---- stage helpers ----
    auto gemm = [&](const u16* Am, const u16* Bw, const float* bias, void* out,
                    int P, int Ktot, int Cout, int CoutPad, int HW, int outmode, int act) {
        int blocks_m = P >> 7;                 // 128 pixels per block (4 waves x 32)
        int n_groups = CoutPad >> 6;           // 64 Cout per group
        size_t lds = (size_t)64 * (Ktot + 8) * sizeof(u16);
        k_gemm<<<blocks_m * n_groups, 128, lds, stream>>>(
            Am, Bw, bias, out, P, Ktot, Cout, CoutPad, HW, outmode, act);
    };
    auto conv = [&](const u16* in1, const u16* in2, int C1, int C2,
                    const u16* Bw, const float* bias, void* out,
                    int outmode, int act, int Hh, int Ww, int Cout, int CoutPad) {
        int P = NB * Hh * Ww, C = C1 + C2;
        int t1 = P * 9;
        k_im2col<<<(t1 + 255) / 256, 256, 0, stream>>>(in1, in2, C1, C2, NB, Hh, Ww, Aim);
        gemm(Aim, Bw, bias, out, P, 9 * C, Cout, CoutPad, Hh * Ww, outmode, act);
    };
    auto mdcn = [&](const u16* xcl, const u16* Bw, const float* bias,
                    void* out, int outmode, int act, int Hh, int Ww) {
        int P = NB * Hh * Ww;
        int t = P * NG * KTAP;
        k_cols<<<(t + 255) / 256, 256, 0, stream>>>(xcl, dcnbuf, NB, Hh, Ww, Aim);
        gemm(Aim, Bw, bias, out, P, 576, MC, MC, Hh * Ww, outmode, act);
    };
    auto up2 = [&](const u16* s, u16* d, int hh, int ww, float scale) {
        int total = NB * 4 * hh * ww * MC;
        k_up2<<<(total + 255) / 256, 256, 0, stream>>>(s, d, NB, hh, ww, MC, scale);
    };

    // ---- level 3 (32x32) ----
    conv(cnf3, crf3, MC, MC, B_oc1[2], oc1_b[2], tmp3, 0, 1, h3, w3, MC, MC);
    conv(tmp3, nullptr, MC, 0, B_oc2[2], oc2_b[2], off3, 0, 1, h3, w3, MC, MC);
    conv(off3, nullptr, MC, 0, B_dow[2], dcn_ob[2], dcnbuf, 1, 0, h3, w3, 216, 256);
    mdcn(cnf3, B_dw[2], dcn_b[2], feat3, 0, 1, h3, w3);            // i==3: lrelu
    up2(off3,  upoff3,  h3, w3, 2.0f);
    up2(feat3, upfeat3, h3, w3, 1.0f);

    // ---- level 2 (64x64) ----
    conv(cnf2, crf2, MC, MC, B_oc1[1], oc1_b[1], t2a, 0, 1, h2, w2, MC, MC);
    conv(t2a, upoff3, MC, MC, B_oc2[1], oc2_b[1], t2b, 0, 1, h2, w2, MC, MC);
    conv(t2b, nullptr, MC, 0, B_oc3[1], oc3_b[1], off2, 0, 1, h2, w2, MC, MC);
    conv(off2, nullptr, MC, 0, B_dow[1], dcn_ob[1], dcnbuf, 1, 0, h2, w2, 216, 256);
    mdcn(cnf2, B_dw[1], dcn_b[1], featd2, 0, 0, h2, w2);           // no act before feat conv
    conv(featd2, upfeat3, MC, MC, B_feat[1], feat_b[1], feat2, 0, 1, h2, w2, MC, MC); // i==2: lrelu
    up2(off2,  upoff2,  h2, w2, 2.0f);
    up2(feat2, upfeat2, h2, w2, 1.0f);

    // ---- level 1 (128x128) ----
    conv(cnf1, crf1, MC, MC, B_oc1[0], oc1_b[0], t1a, 0, 1, h1, w1, MC, MC);
    conv(t1a, upoff2, MC, MC, B_oc2[0], oc2_b[0], t1b, 0, 1, h1, w1, MC, MC);
    conv(t1b, nullptr, MC, 0, B_oc3[0], oc3_b[0], off1, 0, 1, h1, w1, MC, MC);
    conv(off1, nullptr, MC, 0, B_dow[0], dcn_ob[0], dcnbuf, 1, 0, h1, w1, 216, 256);
    mdcn(cnf1, B_dw[0], dcn_b[0], featd1, 0, 0, h1, w1);
    conv(featd1, upfeat2, MC, MC, B_feat[0], feat_b[0], feat1, 0, 0, h1, w1, MC, MC); // i==1: no act

    // ---- cascade ----
    conv(feat1, crf1, MC, MC, B_cas1, cas1_b, tca, 0, 1, h1, w1, MC, MC);
    conv(tca, nullptr, MC, 0, B_cas2, cas2_b, tcb, 0, 1, h1, w1, MC, MC);
    conv(tcb, nullptr, MC, 0, B_cow, cas_ob, dcnbuf, 1, 0, h1, w1, 216, 256);
    mdcn(feat1, B_cw, cas_b, d_out, 2, 1, h1, w1);  // final: lrelu, f32 NCHW into d_out
}
